// MultiHeadAttention_69148973466311
// MI455X (gfx1250) — compile-verified, matrix operand order checked by reading, not run on hardware
//
#include <hip/hip_runtime.h>

// ---------------------------------------------------------------------------
// Types for CDNA5 WMMA (gfx1250, wave32)
// ---------------------------------------------------------------------------
typedef __attribute__((ext_vector_type(16))) __bf16 v16bf;
typedef __attribute__((ext_vector_type(8)))  __bf16 v8bf;
typedef __attribute__((ext_vector_type(8)))  float  v8f;

// Problem constants (from reference)
constexpr int Bn  = 2;
constexpr int Sn  = 2048;
constexpr int Dn  = 1024;
constexpr int Hn  = 16;
constexpr int HDn = 64;
constexpr int Mn  = Bn * Sn;                     // 4096 rows
// softmax in base-2 domain: exp(x/4096) = exp2(x * log2(e)/4096).
// This scale is folded into the Q projection epilogue.
constexpr float SCALE2 = 1.4426950408889634f / 4096.0f;

// Raw v_exp_f32 (base-2). Inputs are <= 0; FTZ below 2^-126 is fine for softmax.
#if __has_builtin(__builtin_amdgcn_exp2f)
#define EXP2F(x) __builtin_amdgcn_exp2f(x)
#else
#define EXP2F(x) exp2f(x)
#endif

// Raw v_max_num_f32: avoids the canonicalize (self-max) ops the backend wraps
// around llvm.maxnum inputs. NaN-as-number semantics are fine for softmax.
__device__ __forceinline__ float vmaxf(float a, float b) {
  float r;
  asm("v_max_num_f32 %0, %1, %2" : "=v"(r) : "v"(a), "v"(b));
  return r;
}

// Raw v_permlane16_b32: 16-lane xor-shuffle in the VALU pipe, no LDS/waits,
// and (unlike the intrinsic) no tied-old destination -> no extra v_mov.
// With EXEC all ones the OLD operand is never consumed (FI=0).
template <unsigned LO, unsigned HI>
__device__ __forceinline__ float lane16_xor(float v) {
  float r;
  asm("v_permlane16_b32 %0, %1, %2, %3" : "=v"(r) : "v"(v), "s"(LO), "s"(HI));
  return r;
}
#define XOR1(v) lane16_xor<0x67452301u, 0xEFCDAB89u>(v)
#define XOR2(v) lane16_xor<0x54761032u, 0xDCFE98BAu>(v)
#define XOR4(v) lane16_xor<0x32107654u, 0xBA98FEDCu>(v)
#define XOR8(v) lane16_xor<0xFEDCBA98u, 0x76543210u>(v)

__device__ __forceinline__ v16bf mk16(v8bf lo, v8bf hi) {
  v16bf r;
#pragma unroll
  for (int i = 0; i < 8; ++i) { r[i] = lo[i]; r[i + 8] = hi[i]; }
  return r;
}

__device__ __forceinline__ v8f wmma_bf16(v16bf a, v16bf b, v8f c) {
  return __builtin_amdgcn_wmma_f32_16x16x32_bf16(
      false, a, false, b, (short)0, c, false, false);
}

// A/B fragment from a row-major 16-wide tile: lane = row (mod 16),
// lane-half selects K sub-ranges {0..7,16..23} vs {8..15,24..31}.
__device__ __forceinline__ v16bf load_frag_rowmajor(const __bf16* __restrict__ base,
                                                    size_t row_stride, int row,
                                                    int k0, int lh) {
  const __bf16* p = base + (size_t)row * row_stride + k0 + lh * 8;
  return mk16(*(const v8bf*)p, *(const v8bf*)(p + 16));
}

// ---------------------------------------------------------------------------
// Casts
// ---------------------------------------------------------------------------
__global__ void mha_cast_bf16(const float* __restrict__ src,
                              __bf16* __restrict__ dst, int n) {
  int i = blockIdx.x * blockDim.x + threadIdx.x;
  if (i < n) dst[i] = (__bf16)src[i];
}

// W[K][N] f32 -> Wt[N][K] bf16
__global__ void mha_castT_bf16(const float* __restrict__ W,
                               __bf16* __restrict__ Wt, int K, int N) {
  int n = blockIdx.x * 16 + threadIdx.x;
  int k = blockIdx.y * 16 + threadIdx.y;
  if (n < N && k < K) Wt[(size_t)n * K + k] = (__bf16)W[(size_t)k * N + n];
}

// ---------------------------------------------------------------------------
// WMMA GEMM:  C[M,N] = A[M,K](bf16 row-major) x Bt[N,K](bf16 = B^T)
// One wave computes a 32x64 strip: 2 A x 4 B fragments -> 8 WMMAs per k-step.
// MODE 0: out bf16 row-major, scaled by oscale   (Q gets log2(e)/hd^2 folded in)
// MODE 1: out bf16 head-transposed V^T[b,h,hd,S]
// MODE 2: out f32 row-major + bias               (output projection)
// ---------------------------------------------------------------------------
template <int MODE>
__global__ void mha_gemm_wmma(const __bf16* __restrict__ A,
                              const __bf16* __restrict__ Bt,
                              void* __restrict__ out,
                              const float* __restrict__ bias,
                              float oscale, int M, int N, int K) {
  const int lane = threadIdx.x & 31;
  const int wave = threadIdx.x >> 5;
  const int t    = blockIdx.x * (blockDim.x >> 5) + wave;
  const int NST  = N >> 6;
  const int mt   = t / NST;
  const int nst  = t % NST;
  if (mt >= (M >> 5)) return;                 // wave-uniform guard
  const int m0 = mt << 5;
  const int n0 = nst << 6;
  const int ln = lane & 15;
  const int lh = lane >> 4;

  v8f acc[2][4] = {};

  for (int k0 = 0; k0 < K; k0 += 32) {
    v16bf af[2];
#pragma unroll
    for (int am = 0; am < 2; ++am) {
      const __bf16* ap = A + (size_t)(m0 + am * 16 + ln) * K + k0 + lh * 8;
      af[am] = mk16(*(const v8bf*)ap, *(const v8bf*)(ap + 16));
      if (k0 + 32 < K) __builtin_prefetch(ap + 32, 0, 0);
    }
#pragma unroll
    for (int nt = 0; nt < 4; ++nt) {
      const __bf16* bp = Bt + (size_t)(n0 + nt * 16 + ln) * K + k0 + lh * 8;
      v16bf bf_ = mk16(*(const v8bf*)bp, *(const v8bf*)(bp + 16));
#pragma unroll
      for (int am = 0; am < 2; ++am)
        acc[am][nt] = wmma_bf16(af[am], bf_, acc[am][nt]);
    }
  }

#pragma unroll
  for (int am = 0; am < 2; ++am)
#pragma unroll
    for (int nt = 0; nt < 4; ++nt)
#pragma unroll
      for (int r = 0; r < 8; ++r) {
        const int m = m0 + am * 16 + r + 8 * lh;
        const int n = n0 + nt * 16 + ln;
        const float v = acc[am][nt][r];
        if (MODE == 0) {
          ((__bf16*)out)[(size_t)m * N + n] = (__bf16)(v * oscale);
        } else if (MODE == 1) {
          const size_t b = (size_t)(m >> 11);
          const int    s = m & (Sn - 1);
          const int    h = n >> 6;
          const int    d = n & (HDn - 1);
          ((__bf16*)out)[((b * Hn + h) * (size_t)HDn + d) * Sn + s] = (__bf16)v;
        } else {
          ((float*)out)[(size_t)m * N + n] = v + bias[n];
        }
      }
}

// ---------------------------------------------------------------------------
// One 16q x 32k attention block. MASK=false below the causal diagonal.
// Scores arrive pre-scaled to the base-2 domain (folded into Q projection).
// Row max: v_permlane16 + v_max_num butterfly (2 instrs/step, no canonicalize).
// Row sums: extra WMMA against a ones B-fragment (1 XDL op instead of ~40 VALU).
// ---------------------------------------------------------------------------
template <bool MASK>
__device__ __forceinline__ void attn_block(
    const __bf16* __restrict__ Kb, const __bf16* __restrict__ Vb,
    const v16bf (&qa)[2], const v16bf& ones, __bf16 (&plds)[16][32],
    float (&rowmax)[8], v8f (&o)[4], v8f& osum,
    int kb0, int q0, int ln, int lh) {
  // ---- scores (two 16-col tiles over 32 keys, hd contraction = 2 k-steps)
  v8f s[2] = {};
#pragma unroll
  for (int nt = 0; nt < 2; ++nt)
#pragma unroll
    for (int f = 0; f < 2; ++f) {
      v16bf kf = load_frag_rowmajor(Kb, Dn, kb0 + nt * 16 + ln, f * 32, lh);
      s[nt] = wmma_bf16(qa[f], kf, s[nt]);
    }

  // ---- causal mask only on the diagonal block
  if (MASK) {
#pragma unroll
    for (int nt = 0; nt < 2; ++nt)
#pragma unroll
      for (int r = 0; r < 8; ++r) {
        const int m = q0 + r + 8 * lh;
        const int n = kb0 + nt * 16 + ln;
        if (n > m) s[nt][r] = -__builtin_inff();
      }
  }

  // ---- online softmax; rows live across the 16-lane half-wave
  float ef[8];
#pragma unroll
  for (int r = 0; r < 8; ++r) {
    float bm = vmaxf(s[0][r], s[1][r]);
    bm = vmaxf(bm, XOR1(bm));
    bm = vmaxf(bm, XOR2(bm));
    bm = vmaxf(bm, XOR4(bm));
    bm = vmaxf(bm, XOR8(bm));
    const float nm = vmaxf(rowmax[r], bm);
    ef[r] = EXP2F(rowmax[r] - nm);
    rowmax[r] = nm;
  }
#pragma unroll
  for (int nt = 0; nt < 2; ++nt)
#pragma unroll
    for (int r = 0; r < 8; ++r)
      s[nt][r] = EXP2F(s[nt][r] - rowmax[r]);

  // ---- rescale running accumulators (O and the row-sum accumulator)
#pragma unroll
  for (int r = 0; r < 8; ++r) {
    osum[r] *= ef[r];
#pragma unroll
    for (int nt = 0; nt < 4; ++nt) o[nt][r] *= ef[r];
  }

  // ---- P: C-layout -> A-layout through the per-wave LDS tile
#pragma unroll
  for (int nt = 0; nt < 2; ++nt)
#pragma unroll
    for (int r = 0; r < 8; ++r)
      plds[r + 8 * lh][nt * 16 + ln] = (__bf16)s[nt][r];
  const __bf16* pr = &plds[ln][lh * 8];
  v16bf pa = mk16(*(const v8bf*)pr, *(const v8bf*)(pr + 16));

  // ---- row sums on the matrix pipe: osum += P x ones
  osum = wmma_bf16(pa, ones, osum);

  // ---- O += P V  (V^T rows contiguous along keys)
#pragma unroll
  for (int nt = 0; nt < 4; ++nt) {
    v16bf vb = load_frag_rowmajor(Vb, Sn, nt * 16 + ln, kb0, lh);
    o[nt] = wmma_bf16(pa, vb, o[nt]);
  }
}

// ---------------------------------------------------------------------------
// Causal flash attention, one wave per (b, h, 16-query tile).
// ---------------------------------------------------------------------------
__global__ void mha_attn_wmma(const __bf16* __restrict__ Qp,
                              const __bf16* __restrict__ Kp,
                              const __bf16* __restrict__ Vt,
                              __bf16* __restrict__ ctx) {
  __shared__ __align__(64) __bf16 plds[4][16][32];

  const int lane = threadIdx.x & 31;
  const int wave = threadIdx.x >> 5;
  const int t    = blockIdx.x * 4 + wave;
  const int QT   = Sn / 16;
  const int qt = t % QT;
  const int h  = (t / QT) % Hn;
  const int b  = t / (QT * Hn);
  const int q0 = qt * 16;
  const int ln = lane & 15;
  const int lh = lane >> 4;

  const __bf16* Qb = Qp + (size_t)b * Sn * Dn + (size_t)h * HDn;
  const __bf16* Kb = Kp + (size_t)b * Sn * Dn + (size_t)h * HDn;
  const __bf16* Vb = Vt + (size_t)(b * Hn + h) * HDn * Sn;

  v16bf qa[2];
#pragma unroll
  for (int f = 0; f < 2; ++f)
    qa[f] = load_frag_rowmajor(Qb, Dn, q0 + ln, f * 32, lh);

  v16bf ones;
#pragma unroll
  for (int i = 0; i < 16; ++i) ones[i] = (__bf16)1.0f;

  float rowmax[8];
#pragma unroll
  for (int r = 0; r < 8; ++r) rowmax[r] = -__builtin_inff();
  v8f o[4] = {};
  v8f osum = {};

  // Blocks with kb0+31 <= q0 are fully unmasked; exactly one diagonal block.
  const int nfull = (q0 + 1) >> 5;
  for (int kb = 0; kb < nfull; ++kb)
    attn_block<false>(Kb, Vb, qa, ones, plds[wave], rowmax, o, osum,
                      kb * 32, q0, ln, lh);
  attn_block<true>(Kb, Vb, qa, ones, plds[wave], rowmax, o, osum,
                   nfull * 32, q0, ln, lh);

  // ---- normalize + store ctx in [B,H,S,hd] ("bug") layout
#pragma unroll
  for (int nt = 0; nt < 4; ++nt)
#pragma unroll
    for (int r = 0; r < 8; ++r) {
      const int m = q0 + r + 8 * lh;
      const int d = nt * 16 + ln;
      const float v = o[nt][r] * (1.0f / osum[r]);
      ctx[((size_t)(b * Hn + h) * Sn + m) * HDn + d] = (__bf16)v;
    }
}

// ---------------------------------------------------------------------------
// Launcher
// ---------------------------------------------------------------------------
extern "C" void kernel_launch(void* const* d_in, const int* in_sizes, int n_in,
                              void* d_out, int out_size, void* d_ws, size_t ws_size,
                              hipStream_t stream) {
  const float* x  = (const float*)d_in[0];
  const float* Wq = (const float*)d_in[1];
  const float* Wk = (const float*)d_in[2];
  const float* Wv = (const float*)d_in[3];
  const float* Wo = (const float*)d_in[4];
  const float* bo = (const float*)d_in[5];
  float* out = (float*)d_out;

  size_t off = 0;
  auto carve = [&](size_t bytes) {
    void* p = (char*)d_ws + off;
    off += (bytes + 255) & ~(size_t)255;
    return p;
  };
  const size_t actBytes = (size_t)Mn * Dn * sizeof(__bf16);   // 8 MiB
  const size_t wBytes   = (size_t)Dn * Dn * sizeof(__bf16);   // 2 MiB
  __bf16* xbf = (__bf16*)carve(actBytes);
  __bf16* WqT = (__bf16*)carve(wBytes);
  __bf16* WkT = (__bf16*)carve(wBytes);
  __bf16* WvT = (__bf16*)carve(wBytes);
  __bf16* WoT = (__bf16*)carve(wBytes);
  __bf16* Qp  = (__bf16*)carve(actBytes);
  __bf16* Kp  = (__bf16*)carve(actBytes);
  __bf16* Vt  = (__bf16*)carve(actBytes);
  __bf16* ctx = (__bf16*)carve(actBytes);
  (void)ws_size; (void)n_in; (void)in_sizes; (void)out_size;

  // 1) casts
  {
    const int n = Mn * Dn;
    mha_cast_bf16<<<(n + 255) / 256, 256, 0, stream>>>(x, xbf, n);
    dim3 tg(16, 16), gg(Dn / 16, Dn / 16);
    mha_castT_bf16<<<gg, tg, 0, stream>>>(Wq, WqT, Dn, Dn);
    mha_castT_bf16<<<gg, tg, 0, stream>>>(Wk, WkT, Dn, Dn);
    mha_castT_bf16<<<gg, tg, 0, stream>>>(Wv, WvT, Dn, Dn);
    mha_castT_bf16<<<gg, tg, 0, stream>>>(Wo, WoT, Dn, Dn);
  }

  // 2) projections: waves = (M/32)*(N/64) = 2048 -> 512 blocks of 4 waves
  const int gemmBlocks = (Mn / 32) * (Dn / 64) / 4;
  mha_gemm_wmma<0><<<gemmBlocks, 128, 0, stream>>>(xbf, WqT, Qp, nullptr, SCALE2, Mn, Dn, Dn);
  mha_gemm_wmma<0><<<gemmBlocks, 128, 0, stream>>>(xbf, WkT, Kp, nullptr, 1.0f,   Mn, Dn, Dn);
  mha_gemm_wmma<1><<<gemmBlocks, 128, 0, stream>>>(xbf, WvT, Vt, nullptr, 1.0f,   Mn, Dn, Dn);

  // 3) attention: B*H*(S/16) = 4096 waves -> 1024 blocks of 4 waves
  mha_attn_wmma<<<Bn * Hn * (Sn / 16) / 4, 128, 0, stream>>>(Qp, Kp, Vt, ctx);

  // 4) output projection with bias (ctx viewed as [4096 x 1024], reshape bug)
  mha_gemm_wmma<2><<<gemmBlocks, 128, 0, stream>>>(ctx, WoT, out, bo, 1.0f, Mn, Dn, Dn);
}
